// BraidedRWKVMamba_2018634629638
// MI455X (gfx1250) — matrix operand order repaired and checked by reading, not compiled, not run on hardware
//
#include <hip/hip_runtime.h>
#include <hip/hip_bf16.h>
#include <math.h>

// ---------------- model constants ----------------
#define DIMC    1024
#define HEADSC  8
#define HDC     128
#define DSTATEC 16
#define DCONVC  4
#define DINNERC 2048
#define BATCHC  2
#define SEQC    1024
#define BTC     (BATCHC*SEQC)
#define VOCABC  32000

// ---------------- vector types -------------------
typedef __attribute__((ext_vector_type(16))) __bf16 v16bf;
typedef __attribute__((ext_vector_type(8)))  __bf16 v8bf;
typedef __attribute__((ext_vector_type(8)))  float  v8f;

__device__ __forceinline__ __bf16 f2bf(float f) {
  unsigned u = __float_as_uint(f);
  u += 0x7fffu + ((u >> 16) & 1u);           // round-to-nearest-even
  unsigned short s = (unsigned short)(u >> 16);
  return __builtin_bit_cast(__bf16, s);
}
__device__ __forceinline__ float siluf(float x) { return x / (1.f + expf(-x)); }

// ---------------- Tensor Data Mover (TDM) ----------------
#if defined(__has_builtin)
#  if __has_builtin(__builtin_amdgcn_tensor_load_to_lds)
#    define HAVE_TDM 1
#  endif
#endif
#ifndef HAVE_TDM
#  define HAVE_TDM 0
#endif

#if HAVE_TDM
typedef __attribute__((ext_vector_type(4))) unsigned tdm_v4u;
typedef __attribute__((ext_vector_type(8))) int      tdm_v8i;
typedef __attribute__((ext_vector_type(4))) int      tdm_v4i;

// 2D tile DMA: global (row-major, 2-byte elems, row stride `stride` elems)
// -> LDS at byte offset lds_off, rows packed contiguously (tile_w*2 bytes per row).
// D# layout per CDNA5 ISA 8.3/8.4 (group0 128b, group1 256b; groups 2/3 zero => <=2D).
// tw/th are remaining tensor extents from the tile start (OOB reads return zero).
__device__ __forceinline__ void tdm_load_2d(unsigned lds_off, const void* gptr,
                                            unsigned tw, unsigned th, unsigned stride,
                                            unsigned tile_w, unsigned tile_h)
{
  unsigned long long ga = (unsigned long long)(uintptr_t)gptr;
  tdm_v4u g0;
  g0[0] = 1u;                                                // count=1, user mode, no gather
  g0[1] = lds_off;                                           // lds_addr [63:32]
  g0[2] = (unsigned)ga;                                      // global_addr[31:0]
  g0[3] = (unsigned)((ga >> 32) & 0x01ffffffull)             // global_addr[56:32]
          | (2u << 30);                                      // type=2 ("image")
  tdm_v8i g1;
  g1[0] = (int)(1u << 16);                                   // wg_mask=0, data_size=1 (2B)
  g1[1] = (int)((tw & 0xffffu) << 16);                       // tensor_dim0[15:0]
  g1[2] = (int)(((tw >> 16) & 0xffffu) | ((th & 0xffffu) << 16));      // td0[31:16] | td1[15:0]
  g1[3] = (int)(((th >> 16) & 0xffffu) | ((tile_w & 0xffffu) << 16));  // td1[31:16] | tile_dim0
  g1[4] = (int)(tile_h & 0xffffu);                           // tile_dim1 (tile_dim2=0)
  g1[5] = (int)stride;                                       // tensor_dim0_stride[31:0]
  g1[6] = 0;                                                 // stride[47:32]=0, dim1_stride lo=0
  g1[7] = 0;
  tdm_v4i z4 = {};
#if __clang_major__ >= 23
  tdm_v8i z8 = {};
  __builtin_amdgcn_tensor_load_to_lds(g0, g1, z4, z4, z8, 0);
#else
  __builtin_amdgcn_tensor_load_to_lds(g0, g1, z4, z4, 0);
#endif
}
#endif // HAVE_TDM

// =================================================
//  bf16 WMMA TN-GEMM:  C[M,N] (f32) = A[M,K] * Bt[N,K]^T + (R)
//  Both operands row-major with K innermost -> both tiles staged by TDM,
//  double-buffered in LDS (DMA of tile i+1 overlaps WMMA on tile i).
//  Block tile 128x128, K-step 32, 8 waves of 64x32 each.
// =================================================
#define BM  128
#define BN  128
#define BKT 32

template<bool RESID>
__global__ __launch_bounds__(256)
void k_gemm_bf16(float* __restrict__ C, const __bf16* __restrict__ A,
                 const __bf16* __restrict__ Bt, const float* __restrict__ R,
                 int M, int N, int K)
{
  __shared__ __bf16 As[2][BM][BKT];   // [buf][m][k]
  __shared__ __bf16 Bs[2][BN][BKT];   // [buf][n][k]

  const int tid  = threadIdx.x;
  const int lane = tid & 31;
  const int wave = tid >> 5;
  const int half = lane >> 4;      // 0: lanes 0-15, 1: lanes 16-31
  const int l16  = lane & 15;
  const int bm = blockIdx.y * BM;
  const int bn = blockIdx.x * BN;
  const int wm = (wave >> 2) * 64; // 2x4 wave grid
  const int wn = (wave & 3) * 32;

  v8f acc[4][2] = {};
  int buf = 0;

#if HAVE_TDM
  const unsigned lds_a0 = (unsigned)(uintptr_t)&As[0][0][0];
  const unsigned lds_b0 = (unsigned)(uintptr_t)&Bs[0][0][0];
  const unsigned abufB  = (unsigned)(BM * BKT * 2);
  const unsigned bbufB  = (unsigned)(BN * BKT * 2);
  // prologue: DMA tile 0 into buffer 0
  if (tid == 0) {
    tdm_load_2d(lds_a0, A  + (size_t)bm * K, (unsigned)K, (unsigned)(M - bm),
                (unsigned)K, BKT, BM);
    tdm_load_2d(lds_b0, Bt + (size_t)bn * K, (unsigned)K, (unsigned)(N - bn),
                (unsigned)K, BKT, BN);
  }
#else
  const int ldr = tid >> 1;        // 0..127
  const int ldc = (tid & 1) * 16;  // 0 or 16
#endif

  for (int k0 = 0; k0 < K; k0 += BKT) {
#if HAVE_TDM
    // wait for this iteration's tiles, publish to all waves
    if (tid == 0) __builtin_amdgcn_s_wait_tensorcnt(0);
    __syncthreads();
    // kick off DMA for the next tiles into the other buffer (overlaps compute)
    if (tid == 0 && (k0 + BKT) < K) {
      unsigned nb = (unsigned)(buf ^ 1);
      tdm_load_2d(lds_a0 + nb * abufB, A  + (size_t)bm * K + k0 + BKT,
                  (unsigned)(K - k0 - BKT), (unsigned)(M - bm), (unsigned)K, BKT, BM);
      tdm_load_2d(lds_b0 + nb * bbufB, Bt + (size_t)bn * K + k0 + BKT,
                  (unsigned)(K - k0 - BKT), (unsigned)(N - bn), (unsigned)K, BKT, BN);
    }
#else
    // ---- manual staging fallback (single buffer, two barriers) ----
    {
      int grow = bm + ldr;
      v8bf z = {};
      if (grow < M && (k0 + ldc) < K) {
        const __bf16* src = A + (size_t)grow * K + k0 + ldc;
        *(v8bf*)&As[0][ldr][ldc]     = *(const v8bf*)(src);
        *(v8bf*)&As[0][ldr][ldc + 8] = *(const v8bf*)(src + 8);
      } else {
        *(v8bf*)&As[0][ldr][ldc]     = z;
        *(v8bf*)&As[0][ldr][ldc + 8] = z;
      }
      int gcol = bn + ldr;
      if (gcol < N && (k0 + ldc) < K) {
        const __bf16* src = Bt + (size_t)gcol * K + k0 + ldc;
        *(v8bf*)&Bs[0][ldr][ldc]     = *(const v8bf*)(src);
        *(v8bf*)&Bs[0][ldr][ldc + 8] = *(const v8bf*)(src + 8);
      } else {
        *(v8bf*)&Bs[0][ldr][ldc]     = z;
        *(v8bf*)&Bs[0][ldr][ldc + 8] = z;
      }
    }
    if (k0 + BKT < K) {
      __builtin_prefetch(A  + (size_t)(bm + ldr) * K + k0 + BKT + ldc, 0, 0);
      __builtin_prefetch(Bt + (size_t)(bn + ldr) * K + k0 + BKT + ldc, 0, 0);
    }
    __syncthreads();
#endif

    // ---- build fragments per ISA 7.12.2 wave32 layouts ----
    v16bf af[4];
    #pragma unroll
    for (int mt = 0; mt < 4; mt++) {
      int mr = wm + mt * 16 + l16;
      // A 16x32 bf16: half0 -> K 0-7 & 16-23, half1 -> K 8-15 & 24-31
      v8bf lo = *(const v8bf*)&As[buf][mr][half * 8];
      v8bf hi = *(const v8bf*)&As[buf][mr][16 + half * 8];
      #pragma unroll
      for (int i = 0; i < 8; i++) { af[mt][i] = lo[i]; af[mt][i + 8] = hi[i]; }
    }
    v16bf bfr[2];
    #pragma unroll
    for (int nt = 0; nt < 2; nt++) {
      int nc = wn + nt * 16 + l16;
      // B 32x16 bf16: half0 -> K 0-7 & 16-23, half1 -> K 8-15 & 24-31 of column (lane&15)
      v8bf lo = *(const v8bf*)&Bs[buf][nc][half * 8];
      v8bf hi = *(const v8bf*)&Bs[buf][nc][16 + half * 8];
      #pragma unroll
      for (int i = 0; i < 8; i++) { bfr[nt][i] = lo[i]; bfr[nt][i + 8] = hi[i]; }
    }
    #pragma unroll
    for (int mt = 0; mt < 4; mt++)
      #pragma unroll
      for (int nt = 0; nt < 2; nt++)
        acc[mt][nt] = __builtin_amdgcn_wmma_f32_16x16x32_bf16(
            false, af[mt], false, bfr[nt], (short)0, acc[mt][nt], false, false);

#if HAVE_TDM
    buf ^= 1;
#else
    __syncthreads();
#endif
  }

  // ---- store: C layout VGPR j -> M = j + 8*half, N = lane&15 ----
  #pragma unroll
  for (int mt = 0; mt < 4; mt++)
    #pragma unroll
    for (int nt = 0; nt < 2; nt++)
      #pragma unroll
      for (int j = 0; j < 8; j++) {
        int row = bm + wm + mt * 16 + half * 8 + j;
        int col = bn + wn + nt * 16 + l16;
        if (row < M && col < N) {
          float v = acc[mt][nt][j];
          if (RESID) v += R[(size_t)row * N + col];
          C[(size_t)row * N + col] = v;
        }
      }
}

// ================= elementwise / recurrent kernels =================

__global__ void k_embed(float* __restrict__ x, const float* __restrict__ emb,
                        const int* __restrict__ tok, int D)
{
  size_t row = blockIdx.x;
  int t = tok[row];
  const float* src = emb + (size_t)t * D;
  float* dst = x + row * D;
  for (int d = threadIdx.x; d < D; d += blockDim.x) dst[d] = src[d];
}

__global__ void k_cvt_bf16(__bf16* __restrict__ dst, const float* __restrict__ src, long long n)
{
  long long stride = (long long)gridDim.x * blockDim.x;
  for (long long i = (long long)blockIdx.x * blockDim.x + threadIdx.x; i < n; i += stride)
    dst[i] = f2bf(src[i]);
}

// src is [K][N] f32 row-major; dst is [N][K] bf16 row-major (transpose + convert)
__global__ void k_cvtT_bf16(__bf16* __restrict__ dst, const float* __restrict__ src,
                            int K, int N, long long n)
{
  long long stride = (long long)gridDim.x * blockDim.x;
  for (long long i = (long long)blockIdx.x * blockDim.x + threadIdx.x; i < n; i += stride) {
    int k = (int)(i % K);
    long long nn = i / K;
    dst[i] = f2bf(src[(size_t)k * N + nn]);
  }
}

__global__ void k_silu_bf16(__bf16* __restrict__ dst, const float* __restrict__ src, long long n)
{
  long long stride = (long long)gridDim.x * blockDim.x;
  for (long long i = (long long)blockIdx.x * blockDim.x + threadIdx.x; i < n; i += stride)
    dst[i] = f2bf(siluf(src[i]));
}

__global__ void k_timemix_bf16(__bf16* __restrict__ dst, const float* __restrict__ src,
                               const float* __restrict__ tm, long long n, int D)
{
  long long stride = (long long)gridDim.x * blockDim.x;
  for (long long i = (long long)blockIdx.x * blockDim.x + threadIdx.x; i < n; i += stride)
    dst[i] = f2bf(src[i] * tm[(int)(i % D)]);
}

__global__ __launch_bounds__(256)
void k_layernorm_bf16(__bf16* __restrict__ dst, const float* __restrict__ src,
                      const float* __restrict__ g, const float* __restrict__ b, int D)
{
  __shared__ float s1[256], s2[256];
  size_t row = blockIdx.x;
  const float* xr = src + row * D;
  float s = 0.f, ss = 0.f;
  for (int d = threadIdx.x; d < D; d += 256) { float v = xr[d]; s += v; ss += v * v; }
  s1[threadIdx.x] = s; s2[threadIdx.x] = ss;
  __syncthreads();
  for (int off = 128; off > 0; off >>= 1) {
    if ((int)threadIdx.x < off) {
      s1[threadIdx.x] += s1[threadIdx.x + off];
      s2[threadIdx.x] += s2[threadIdx.x + off];
    }
    __syncthreads();
  }
  float mean = s1[0] / D;
  float var  = s2[0] / D - mean * mean;
  float inv  = rsqrtf(var + 1e-5f);
  __bf16* dr = dst + row * D;
  for (int d = threadIdx.x; d < D; d += 256)
    dr[d] = f2bf((xr[d] - mean) * inv * g[d] + b[d]);
}

// RWKV wkv recurrence: one block per (b,h). State s[k][v] 128x128 in regs
// (thread owns column v = tid&127, half of k range selected by tid>>7).
__global__ __launch_bounds__(256)
void k_rwkv_scan(float* __restrict__ wkv, const float* __restrict__ kmat,
                 const float* __restrict__ vmat, const float* __restrict__ td, int T)
{
  __shared__ float kbuf[HDC], vbuf[HDC], pbuf[256];
  int b = blockIdx.x >> 3, h = blockIdx.x & 7;
  int tid  = threadIdx.x;
  int vcol = tid & (HDC - 1);
  int kh   = tid >> 7;                    // 0 or 1: k range [kh*64, kh*64+64)
  float st[64], dec[64];
  #pragma unroll
  for (int i = 0; i < 64; i++) { st[i] = 0.f; dec[i] = expf(-td[h * HDC + kh * 64 + i]); }
  size_t basebh = (size_t)b * T * DIMC + (size_t)h * HDC;
  for (int t = 0; t < T; t++) {
    size_t base = basebh + (size_t)t * DIMC;
    if (tid < HDC) kbuf[tid] = kmat[base + tid];
    else           vbuf[tid - HDC] = vmat[base + (tid - HDC)];
    __syncthreads();
    float myv = vbuf[vcol];
    float acc = 0.f;
    #pragma unroll
    for (int i = 0; i < 64; i++) {
      float kk = kbuf[kh * 64 + i];
      float s2 = st[i] * dec[i] + kk * myv;   // s = s*decay + k (x) v
      st[i] = s2;
      acc += s2 * kk;                          // out_v = sum_k s[k,v]*k[k]
    }
    pbuf[tid] = acc;
    __syncthreads();
    if (kh == 0) wkv[base + vcol] = pbuf[tid] + pbuf[tid + 128];
    __syncthreads();
  }
}

// out = bf16( r * (wkv + time_first*k*v) * silu(g) )
__global__ void k_rwkv_mix(__bf16* __restrict__ dst, const float* __restrict__ rmat,
                           const float* __restrict__ wkv, const float* __restrict__ gmat,
                           const float* __restrict__ kmat, const float* __restrict__ vmat,
                           const float* __restrict__ tf, long long n, int D)
{
  long long stride = (long long)gridDim.x * blockDim.x;
  for (long long i = (long long)blockIdx.x * blockDim.x + threadIdx.x; i < n; i += stride) {
    int col = (int)(i % D);
    float wt = wkv[i] + tf[col] * kmat[i] * vmat[i];
    dst[i] = f2bf(rmat[i] * wt * siluf(gmat[i]));
  }
}

// causal depthwise conv (DCONV=4) + silu
__global__ void k_conv_silu(float* __restrict__ out, const float* __restrict__ xin,
                            const float* __restrict__ w, int T, int C, long long n)
{
  long long stride = (long long)gridDim.x * blockDim.x;
  for (long long i = (long long)blockIdx.x * blockDim.x + threadIdx.x; i < n; i += stride) {
    int c = (int)(i % C);
    long long bt = i / C;
    int t = (int)(bt % T);
    long long brow = bt - t;                 // b*T
    float acc = 0.f;
    #pragma unroll
    for (int j = 0; j < DCONVC; j++) {
      int tt = t - (DCONVC - 1) + j;
      if (tt >= 0) acc += xin[(brow + tt) * C + c] * w[c * DCONVC + j];
    }
    out[i] = siluf(acc);
  }
}

// dt = softplus(proj[...,0]*Wdt + bdt)
__global__ void k_dtproj(float* __restrict__ dt, const float* __restrict__ proj,
                         const float* __restrict__ Wdt, const float* __restrict__ bdt,
                         long long n, int Dinner)
{
  long long stride = (long long)gridDim.x * blockDim.x;
  for (long long i = (long long)blockIdx.x * blockDim.x + threadIdx.x; i < n; i += stride) {
    int d = (int)(i % Dinner);
    long long bt = i / Dinner;
    float z = proj[bt * 33] * Wdt[d] + bdt[d];
    dt[i] = (z > 20.f) ? z : log1pf(expf(z));
  }
}

// Mamba selective scan: 1 thread per (b, channel), 16-wide state in regs
__global__ __launch_bounds__(256)
void k_mamba_scan(float* __restrict__ ys, const float* __restrict__ dt,
                  const float* __restrict__ proj, const float* __restrict__ xc,
                  const float* __restrict__ A_log, const float* __restrict__ Dp, int T)
{
  __shared__ float Bbuf[DSTATEC], Cbuf[DSTATEC];
  const int blocksPerB = DINNERC / 256;      // 8
  int b = blockIdx.x / blocksPerB;
  int d = (blockIdx.x % blocksPerB) * 256 + threadIdx.x;
  float a[DSTATEC], st[DSTATEC];
  #pragma unroll
  for (int s = 0; s < DSTATEC; s++) { a[s] = -expf(A_log[d * DSTATEC + s]); st[s] = 0.f; }
  float Dd = Dp[d];
  for (int t = 0; t < T; t++) {
    long long bt = (long long)b * T + t;
    if ((int)threadIdx.x < 2 * DSTATEC) {
      float pv = proj[bt * 33 + 1 + threadIdx.x];
      if ((int)threadIdx.x < DSTATEC) Bbuf[threadIdx.x] = pv;
      else                            Cbuf[threadIdx.x - DSTATEC] = pv;
    }
    __syncthreads();
    size_t idx = (size_t)bt * DINNERC + d;
    float xt = xc[idx], dtt = dt[idx];
    float y = 0.f;
    #pragma unroll
    for (int s = 0; s < DSTATEC; s++) {
      float da = expf(a[s] * dtt);
      st[s] = da * st[s] + Bbuf[s] * xt;
      y += st[s] * Cbuf[s];
    }
    ys[idx] = y + Dd * xt;
    __syncthreads();
  }
}

// ============================= host =============================

struct MambaP {
  const float *ln_g, *ln_b, *Win, *conv_w, *Wx, *Wdt, *bdt, *A_log, *D, *Wout;
};

extern "C" void kernel_launch(void* const* d_in, const int* in_sizes, int n_in,
                              void* d_out, int out_size, void* d_ws, size_t ws_size,
                              hipStream_t stream)
{
  (void)in_sizes; (void)n_in; (void)out_size; (void)ws_size;

  // --- input pointers (setup_inputs dict order) ---
  const float* embed = (const float*)d_in[0];
  // layer 0 (rwkv): 1..8 ; layer 1 (mamba): 9..18 ; layer 2 (rwkv): 19..26 ; layer 3 (mamba): 27..36
  const float* R0_tm = (const float*)d_in[1];
  const float* R0_td = (const float*)d_in[2];
  const float* R0_tf = (const float*)d_in[3];
  const float* R0_Wk = (const float*)d_in[4];
  const float* R0_Wv = (const float*)d_in[5];
  const float* R0_Wr = (const float*)d_in[6];
  const float* R0_Wg = (const float*)d_in[7];
  const float* R0_Wo = (const float*)d_in[8];
  MambaP M1 { (const float*)d_in[9],  (const float*)d_in[10], (const float*)d_in[11],
              (const float*)d_in[12], (const float*)d_in[13], (const float*)d_in[14],
              (const float*)d_in[15], (const float*)d_in[16], (const float*)d_in[17],
              (const float*)d_in[18] };
  const float* R2_tm = (const float*)d_in[19];
  const float* R2_td = (const float*)d_in[20];
  const float* R2_tf = (const float*)d_in[21];
  const float* R2_Wk = (const float*)d_in[22];
  const float* R2_Wv = (const float*)d_in[23];
  const float* R2_Wr = (const float*)d_in[24];
  const float* R2_Wg = (const float*)d_in[25];
  const float* R2_Wo = (const float*)d_in[26];
  MambaP M3 { (const float*)d_in[27], (const float*)d_in[28], (const float*)d_in[29],
              (const float*)d_in[30], (const float*)d_in[31], (const float*)d_in[32],
              (const float*)d_in[33], (const float*)d_in[34], (const float*)d_in[35],
              (const float*)d_in[36] };
  const float* norm_g = (const float*)d_in[37];
  const float* norm_b = (const float*)d_in[38];
  const int*   tokens = (const int*)d_in[39];

  // --- workspace carve-up (256B aligned) ---
  char* ws = (char*)d_ws;
  size_t off = 0;
  auto alloc = [&](size_t bytes) -> void* {
    void* p = ws + off;
    off += (bytes + 255) & ~(size_t)255;
    return p;
  };
  float*  f_x      = (float*) alloc((size_t)BTC * DIMC * 4);     // activation (persistent)
  __bf16* bf_dim   = (__bf16*)alloc((size_t)BTC * DIMC * 2);     // bf16 DIM-wide GEMM input
  float*  f_k      = (float*) alloc((size_t)BTC * DIMC * 4);
  float*  f_v      = (float*) alloc((size_t)BTC * DIMC * 4);
  float*  f_r      = (float*) alloc((size_t)BTC * DIMC * 4);
  float*  f_g      = (float*) alloc((size_t)BTC * DIMC * 4);
  float*  f_wkv    = (float*) alloc((size_t)BTC * DIMC * 4);
  float*  f_in1    = (float*) alloc((size_t)BTC * DINNERC * 4);  // xin, later ys
  float*  f_in2    = (float*) alloc((size_t)BTC * DINNERC * 4);  // xconv
  float*  f_dt     = (float*) alloc((size_t)BTC * DINNERC * 4);
  __bf16* bf_inner = (__bf16*)alloc((size_t)BTC * DINNERC * 2);  // bf16 2048-wide GEMM input
  float*  f_proj   = (float*) alloc((size_t)BTC * 33 * 4);
  __bf16* bf_w     = (__bf16*)alloc((size_t)DINNERC * DIMC * 2); // staged bf16 [N][K] weight
  __bf16* bf_emb   = (__bf16*)alloc((size_t)VOCABC * DIMC * 2);  // embed already [N][K]

  dim3 blk(256);
  auto grid1 = [&](long long n) {
    long long b = (n + 255) / 256;
    if (b > 4096) b = 4096;
    return dim3((unsigned)b);
  };
  auto cvt = [&](__bf16* dst, const float* src, long long n) {
    k_cvt_bf16<<<grid1(n), blk, 0, stream>>>(dst, src, n);
  };
  // weight [K][N] f32 -> [N][K] bf16 so every GEMM is TN (TDM-stageable)
  auto cvtT = [&](__bf16* dst, const float* src, int K, int N) {
    long long n = (long long)K * N;
    k_cvtT_bf16<<<grid1(n), blk, 0, stream>>>(dst, src, K, N, n);
  };
  auto gemm = [&](float* C, const __bf16* A, const __bf16* Bt, const float* R,
                  int M, int N, int K) {
    dim3 g((N + BN - 1) / BN, (M + BM - 1) / BM);
    if (R) k_gemm_bf16<true ><<<g, blk, 0, stream>>>(C, A, Bt, R,       M, N, K);
    else   k_gemm_bf16<false><<<g, blk, 0, stream>>>(C, A, Bt, nullptr, M, N, K);
  };

  const long long nD = (long long)BTC * DIMC;
  const long long nI = (long long)BTC * DINNERC;

  auto run_rwkv = [&](const float* tm, const float* td, const float* tf,
                      const float* Wk, const float* Wv, const float* Wr,
                      const float* Wg, const float* Wo) {
    k_timemix_bf16<<<grid1(nD), blk, 0, stream>>>(bf_dim, f_x, tm, nD, DIMC);
    cvtT(bf_w, Wk, DIMC, DIMC);
    gemm(f_k, bf_dim, bf_w, nullptr, BTC, DIMC, DIMC);
    cvtT(bf_w, Wv, DIMC, DIMC);
    gemm(f_v, bf_dim, bf_w, nullptr, BTC, DIMC, DIMC);
    cvtT(bf_w, Wr, DIMC, DIMC);
    gemm(f_r, bf_dim, bf_w, nullptr, BTC, DIMC, DIMC);
    cvtT(bf_w, Wg, DIMC, DIMC);
    gemm(f_g, bf_dim, bf_w, nullptr, BTC, DIMC, DIMC);
    k_rwkv_scan<<<dim3(BATCHC * HEADSC), blk, 0, stream>>>(f_wkv, f_k, f_v, td, SEQC);
    k_rwkv_mix<<<grid1(nD), blk, 0, stream>>>(bf_dim, f_r, f_wkv, f_g, f_k, f_v, tf, nD, DIMC);
    cvtT(bf_w, Wo, DIMC, DIMC);
    gemm(f_x, bf_dim, bf_w, nullptr, BTC, DIMC, DIMC);          // replaces x (no residual)
  };

  auto run_mamba = [&](const MambaP& p) {
    k_layernorm_bf16<<<dim3(BTC), blk, 0, stream>>>(bf_dim, f_x, p.ln_g, p.ln_b, DIMC);
    cvtT(bf_w, p.Win, DIMC, DINNERC);
    gemm(f_in1, bf_dim, bf_w, nullptr, BTC, DINNERC, DIMC);
    k_conv_silu<<<grid1(nI), blk, 0, stream>>>(f_in2, f_in1, p.conv_w, SEQC, DINNERC, nI);
    cvt(bf_inner, f_in2, nI);
    cvtT(bf_w, p.Wx, DINNERC, 33);
    gemm(f_proj, bf_inner, bf_w, nullptr, BTC, 33, DINNERC);
    k_dtproj<<<grid1(nI), blk, 0, stream>>>(f_dt, f_proj, p.Wdt, p.bdt, nI, DINNERC);
    k_mamba_scan<<<dim3(BATCHC * (DINNERC / 256)), blk, 0, stream>>>(
        f_in1, f_dt, f_proj, f_in2, p.A_log, p.D, SEQC);        // ys -> f_in1 (xin dead)
    k_silu_bf16<<<grid1(nI), blk, 0, stream>>>(bf_inner, f_in1, nI);
    cvtT(bf_w, p.Wout, DINNERC, DIMC);
    gemm(f_x, bf_inner, bf_w, f_x, BTC, DIMC, DINNERC);         // + shortcut
  };

  // ---------------- pipeline ----------------
  k_embed<<<dim3(BTC), blk, 0, stream>>>(f_x, embed, tokens, DIMC);
  cvt(bf_emb, embed, (long long)VOCABC * DIMC);

  run_rwkv(R0_tm, R0_td, R0_tf, R0_Wk, R0_Wv, R0_Wr, R0_Wg, R0_Wo);
  run_mamba(M1);
  run_rwkv(R2_tm, R2_td, R2_tf, R2_Wk, R2_Wv, R2_Wr, R2_Wg, R2_Wo);
  run_mamba(M3);

  k_layernorm_bf16<<<dim3(BTC), blk, 0, stream>>>(bf_dim, f_x, norm_g, norm_b, DIMC);
  // logits = ln(x) @ embed^T   (embed is already [VOCAB][DIM] = [N][K])
  gemm((float*)d_out, bf_dim, bf_emb, nullptr, BTC, VOCABC, DIMC);
}
